// AdaptiveDeformConv1d_86629490360666
// MI455X (gfx1250) — matrix operand order repaired and checked by reading, not compiled, not run on hardware
//
#include <hip/hip_runtime.h>
#include <hip/hip_bf16.h>
#include <stdint.h>

#define N_ 2
#define L_ 2048
#define C_ 256
#define G_ 4
#define GC_ 64
#define K_ 31
#define HID_ 32
#define NL_ (N_*L_)
#define NLC_ (N_*L_*C_)

typedef _Float16 v16h __attribute__((ext_vector_type(16)));
typedef _Float16 v8h  __attribute__((ext_vector_type(8)));
typedef float    v8f  __attribute__((ext_vector_type(8)));

// ---------------- small utility kernels ----------------

__global__ void k_zero(float* p, int n){
  int i = blockIdx.x*blockDim.x + threadIdx.x;
  if (i < n) p[i] = 0.f;
}

__global__ void k_cvt(const float* __restrict__ in, _Float16* __restrict__ out, int n){
  int i = blockIdx.x*blockDim.x + threadIdx.x;
  if (i < n) out[i] = (_Float16)in[i];
}

// pack [off_w ; pad ; msk_w ; pad] into a 256x256 f16 weight (rows = output cols)
__global__ void k_cat(const float* __restrict__ offw, const float* __restrict__ offb,
                      const float* __restrict__ mskw, const float* __restrict__ mskb,
                      _Float16* __restrict__ catw, float* __restrict__ catb){
  int i = blockIdx.x*blockDim.x + threadIdx.x;
  if (i < 256*256){
    int r = i >> 8, c = i & 255;
    float v = 0.f;
    if (r < G_*K_)                      v = offw[r*C_ + c];
    else if (r >= 128 && r < 128+G_*K_) v = mskw[(r-128)*C_ + c];
    catw[i] = (_Float16)v;
    if (c == 0){
      float b = 0.f;
      if (r < G_*K_)                      b = offb[r];
      else if (r >= 128 && r < 128+G_*K_) b = mskb[r-128];
      catb[r] = b;
    }
  }
}

// ---------------- SIREN kernel generator + envelope ----------------

__global__ void k_siren(const float* __restrict__ log_sigma,
                        const float* __restrict__ w1, const float* __restrict__ b1,
                        const float* __restrict__ w2, const float* __restrict__ b2,
                        const float* __restrict__ wo, const float* __restrict__ bo,
                        float* __restrict__ kw, float* __restrict__ envo){
  __shared__ float h2s[K_][HID_];
  __shared__ float es[K_];
  __shared__ float esum;
  int t = threadIdx.x;                      // 256 threads
  if (t < K_){
    float grid = -0.5f + (float)t * (1.0f/30.0f);
    float pp = grid * 2.0f;
    float h1[HID_];
    #pragma unroll
    for (int i = 0; i < HID_; ++i) h1[i] = sinf(30.0f*(pp*w1[i] + b1[i]));
    for (int j = 0; j < HID_; ++j){
      float s = b2[j];
      #pragma unroll
      for (int i = 0; i < HID_; ++i) s += h1[i]*w2[j*HID_ + i];
      h2s[t][j] = sinf(30.0f*s);
    }
    float sig = fminf(fmaxf(expf(log_sigma[0]), 0.05f), 0.5f);
    es[t] = expf(-grid*grid/(2.0f*sig*sig));
  }
  __syncthreads();
  if (t == 0){ float s = 0.f; for (int i = 0; i < K_; ++i) s += es[i]; esum = s; }
  __syncthreads();
  if (t < K_) envo[t] = es[t] / esum;
  for (int k = 0; k < K_; ++k){
    float s = bo[t];
    const float* wr = wo + t*HID_;
    #pragma unroll
    for (int j = 0; j < HID_; ++j) s += h2s[k][j]*wr[j];
    kw[t*K_ + k] = s;                       // kw[(g*GC+c)*K + k]
  }
}

// ---------------- depthwise conv (k=3) + LayerNorm + GELU ----------------

__global__ void k_dwln(const float* __restrict__ x, const float* __restrict__ dww,
                       const float* __restrict__ dwb, const float* __restrict__ lng,
                       const float* __restrict__ lnb, _Float16* __restrict__ xdwh){
  int row = blockIdx.x;                     // (n,l)
  int n = row / L_, l = row % L_;
  int c = threadIdx.x;                      // 256 = one channel each
  __shared__ float red[C_];
  __shared__ float mus, vas;
  float v = dwb[c];
  #pragma unroll
  for (int j = 0; j < 3; ++j){
    int ll = l - 1 + j;
    if (ll >= 0 && ll < L_) v += x[(size_t)(n*L_+ll)*C_ + c] * dww[c*3 + j];
  }
  red[c] = v; __syncthreads();
  for (int s = 128; s > 0; s >>= 1){ if (c < s) red[c] += red[c+s]; __syncthreads(); }
  if (c == 0) mus = red[0] * (1.0f/C_);
  __syncthreads();
  float d = v - mus;
  red[c] = d*d; __syncthreads();
  for (int s = 128; s > 0; s >>= 1){ if (c < s) red[c] += red[c+s]; __syncthreads(); }
  if (c == 0) vas = red[0] * (1.0f/C_);
  __syncthreads();
  float y = d * rsqrtf(vas + 1e-5f) * lng[c] + lnb[c];
  float gl = 0.5f*y*(1.0f + tanhf(0.7978845608028654f*(y + 0.044715f*y*y*y)));
  xdwh[(size_t)row*C_ + c] = (_Float16)gl;
}

// ---------------- WMMA GEMM: Out[M,Nc] = A[M,Kd] * W[Nc,Kd]^T + bias ----------------
// One wave per 16x16 tile; K stepped by 32 via v_wmma_f32_16x16x32_f16.

__global__ void k_gemm_wmma(const _Float16* __restrict__ A, const _Float16* __restrict__ W,
                            const float* __restrict__ bias, float* __restrict__ Out,
                            int M, int Nc, int Kd){
  int wave = (blockIdx.x * blockDim.x + threadIdx.x) >> 5;
  int lane = threadIdx.x & 31;
  int tilesN = Nc >> 4;
  int tm = wave / tilesN, tn = wave - tm*tilesN;
  if (tm*16 >= M) return;
  int hl  = lane >> 4;                      // lane half: 0 or 1
  int l16 = lane & 15;
  const _Float16* arow = A + (size_t)(tm*16 + l16) * Kd;
  const _Float16* wrow = W + (size_t)(tn*16 + l16) * Kd;
  v8f acc = {};
  for (int k0 = 0; k0 < Kd; k0 += 32){
    // A frag (16x32 f16): lanes 0-15 hold K {0..7,16..23}, lanes 16-31 hold K {8..15,24..31}
    union { v16h v; v8h h[2]; } a, b;
    a.h[0] = *(const v8h*)(arow + k0 +      hl*8);
    a.h[1] = *(const v8h*)(arow + k0 + 16 + hl*8);
    // B frag (32x16 f16): lane group gives N, same K order as A: {0..7,16..23} / {8..15,24..31}
    b.h[0] = *(const v8h*)(wrow + k0 +      hl*8);
    b.h[1] = *(const v8h*)(wrow + k0 + 16 + hl*8);
    acc = __builtin_amdgcn_wmma_f32_16x16x32_f16(false, a.v, false, b.v,
                                                 (short)0, acc, false, false);
  }
  float bv = bias[tn*16 + l16];
  int m0 = tm*16 + hl*8;                    // D: VGPR r -> M = r (+8 for lanes 16-31)
  size_t nc = (size_t)(tn*16 + l16);
  #pragma unroll
  for (int r = 0; r < 8; ++r)
    Out[(size_t)(m0 + r) * Nc + nc] = acc[r] + bv;
}

// ---------------- offsets finalize + masked softmax + regularizer partials ----------------

__global__ void k_softmax(const float* __restrict__ raw, const float* __restrict__ env,
                          float* __restrict__ offs, float* __restrict__ attn,
                          float* __restrict__ accs){
  int row = blockIdx.x;
  int t = threadIdx.x;                      // 128 = 4 groups x 32
  int g = t >> 5, k = t & 31;
  const float* r = raw + (size_t)row*C_;
  __shared__ float sm[128];
  __shared__ float gmax[G_], gsum[G_];
  bool act = (k < K_);
  float offv = 0.f, m = -1e30f;
  if (act){
    offv = r[g*K_ + k] * 2.0f;              // OFFSET_SCALE
    offs[(size_t)row*(G_*K_) + g*K_ + k] = offv;
    m = r[128 + g*K_ + k] * env[k];
  }
  sm[t] = m; __syncthreads();
  if (k == 0){ float mx = -1e30f; for (int i = 0; i < K_; ++i) mx = fmaxf(mx, sm[g*32+i]); gmax[g] = mx; }
  __syncthreads();
  float e = act ? expf(m - gmax[g]) : 0.f;
  sm[t] = e; __syncthreads();
  if (k == 0){ float s = 0.f; for (int i = 0; i < K_; ++i) s += sm[g*32+i]; gsum[g] = s; }
  __syncthreads();
  float a = act ? e / gsum[g] : 0.f;
  if (!(a == a) || fabsf(a) > 1e37f) a = 0.f;   // nan_to_num
  if (act) attn[(size_t)row*(G_*K_) + g*K_ + k] = a;
  float ent = act ? -a * logf(a + 1e-8f) : 0.f;
  __syncthreads();
  sm[t] = act ? offv*offv : 0.f; __syncthreads();
  if (t == 0){ float s = 0.f; for (int i = 0; i < 128; ++i) s += sm[i]; atomicAdd(&accs[0], s); }
  __syncthreads();
  sm[t] = ent; __syncthreads();
  if (t == 0){ float s = 0.f; for (int i = 0; i < 128; ++i) s += sm[i]; atomicAdd(&accs[1], s); }
}

// ---------------- deformable bilinear gather + einsum -> f16 for out-proj GEMM ----------------

__global__ void k_deform(const float* __restrict__ xproj, const float* __restrict__ offs,
                         const float* __restrict__ attn, const float* __restrict__ kw,
                         _Float16* __restrict__ gouth){
  int row = blockIdx.x;                     // (n,l)
  int n = row / L_, l = row % L_;
  int t = threadIdx.x;                      // 256: g = t/64, c = t%64
  __shared__ float as[G_*K_], os[G_*K_];
  if (t < G_*K_){
    as[t] = attn[(size_t)row*(G_*K_) + t];
    os[t] = offs[(size_t)row*(G_*K_) + t];
  }
  __syncthreads();
  int g = t >> 6, c = t & 63;
  const float* kwr = kw + (size_t)(g*GC_ + c)*K_;
  const float* xb  = xproj + (size_t)n*L_*C_ + g*GC_ + c;
  float acc = 0.f;
  for (int k = 0; k < K_; ++k){
    float a = as[g*K_ + k];
    float s = (float)l + (float)(k - 15) + os[g*K_ + k];
    float i0f = floorf(s);
    float fr  = s - i0f;
    int i0  = (int)i0f;
    int i0c = min(max(i0,   0), L_-1);
    int i1c = min(max(i0+1, 0), L_-1);
    float x0 = xb[(size_t)i0c * C_];        // 64 lanes -> 256B coalesced
    float x1 = xb[(size_t)i1c * C_];
    acc += a * kwr[k] * (x0 + fr*(x1 - x0));
  }
  gouth[(size_t)row*C_ + t] = (_Float16)acc;
}

// ---------------- SE block ----------------

__global__ void k_semean(const float* __restrict__ outp, float* __restrict__ colmean){
  int n = blockIdx.x, c = threadIdx.x;
  const float* p = outp + (size_t)n*L_*C_ + c;
  float s = 0.f;
  for (int l = 0; l < L_; ++l) s += p[(size_t)l*C_];
  colmean[n*C_ + c] = s * (1.0f/L_);
}

__global__ void k_semlp(const float* __restrict__ colmean, const float* __restrict__ w1,
                        const float* __restrict__ b1, const float* __restrict__ w2,
                        const float* __restrict__ b2, float* __restrict__ sew){
  int n = blockIdx.x, t = threadIdx.x;      // 256
  __shared__ float hb[64];
  if (t < 64){
    float s = b1[t];
    const float* wr = w1 + t*C_;
    const float* cm = colmean + n*C_;
    for (int i = 0; i < C_; ++i) s += cm[i]*wr[i];
    hb[t] = s / (1.0f + expf(-s));          // silu
  }
  __syncthreads();
  float u = b2[t];
  const float* w2r = w2 + t*64;
  #pragma unroll
  for (int j = 0; j < 64; ++j) u += hb[j]*w2r[j];
  sew[n*C_ + t] = 1.0f/(1.0f + expf(-u));   // sigmoid
}

__global__ void k_sescale(float* __restrict__ outp, const float* __restrict__ sew){
  int i = blockIdx.x*blockDim.x + threadIdx.x;
  if (i < NLC_){
    int c = i & 255;
    int n = i >> 19;                        // L_*C_ = 2^19
    outp[i] *= sew[n*C_ + c];
  }
}

__global__ void k_scalars(const float* __restrict__ accs, float* __restrict__ outp){
  if (blockIdx.x == 0 && threadIdx.x == 0){
    outp[NLC_]     = accs[0] / (float)(N_*L_*G_*K_);
    outp[NLC_ + 1] = accs[1] / (float)(N_*L_*G_);
  }
}

// ---------------- launch ----------------

extern "C" void kernel_launch(void* const* d_in, const int* in_sizes, int n_in,
                              void* d_out, int out_size, void* d_ws, size_t ws_size,
                              hipStream_t stream){
  const float* x    = (const float*)d_in[0];
  const float* ls   = (const float*)d_in[1];
  const float* knw1 = (const float*)d_in[2];
  const float* knb1 = (const float*)d_in[3];
  const float* knw2 = (const float*)d_in[4];
  const float* knb2 = (const float*)d_in[5];
  const float* knwo = (const float*)d_in[6];
  const float* knbo = (const float*)d_in[7];
  const float* dww  = (const float*)d_in[8];
  const float* dwb  = (const float*)d_in[9];
  const float* lng  = (const float*)d_in[10];
  const float* lnb  = (const float*)d_in[11];
  const float* offw = (const float*)d_in[12];
  const float* offb = (const float*)d_in[13];
  const float* mskw = (const float*)d_in[14];
  const float* mskb = (const float*)d_in[15];
  const float* inw  = (const float*)d_in[16];
  const float* inb  = (const float*)d_in[17];
  const float* outw = (const float*)d_in[18];
  const float* outb = (const float*)d_in[19];
  const float* sew1 = (const float*)d_in[20];
  const float* seb1 = (const float*)d_in[21];
  const float* sew2 = (const float*)d_in[22];
  const float* seb2 = (const float*)d_in[23];
  float* out = (float*)d_out;
  (void)in_sizes; (void)n_in; (void)out_size; (void)ws_size;

  char* p = (char*)d_ws;
  auto take = [&](size_t bytes)->char* {
    uintptr_t q = ((uintptr_t)p + 255) & ~(uintptr_t)255;
    p = (char*)(q + bytes);
    return (char*)q;
  };
  _Float16* x_h    = (_Float16*)take((size_t)NLC_*2);
  _Float16* xdw_h  = (_Float16*)take((size_t)NLC_*2);
  _Float16* gout_h = (_Float16*)take((size_t)NLC_*2);
  _Float16* inw_h  = (_Float16*)take((size_t)C_*C_*2);
  _Float16* outw_h = (_Float16*)take((size_t)C_*C_*2);
  _Float16* catw_h = (_Float16*)take((size_t)C_*C_*2);
  float* xproj   = (float*)take((size_t)NLC_*4);
  float* rawom   = (float*)take((size_t)NL_*C_*4);
  float* offsb   = (float*)take((size_t)NL_*G_*K_*4);
  float* attnb   = (float*)take((size_t)NL_*G_*K_*4);
  float* kw      = (float*)take((size_t)C_*K_*4);
  float* env     = (float*)take((size_t)K_*4);
  float* catb    = (float*)take((size_t)C_*4);
  float* colmean = (float*)take((size_t)N_*C_*4);
  float* sew     = (float*)take((size_t)N_*C_*4);
  float* accs    = (float*)take(2*4);

  k_zero   <<<1, 32, 0, stream>>>(accs, 2);
  k_cvt    <<<(NLC_+255)/256,   256, 0, stream>>>(x,    x_h,    NLC_);
  k_cvt    <<<(C_*C_+255)/256,  256, 0, stream>>>(inw,  inw_h,  C_*C_);
  k_cvt    <<<(C_*C_+255)/256,  256, 0, stream>>>(outw, outw_h, C_*C_);
  k_cat    <<<(C_*C_+255)/256,  256, 0, stream>>>(offw, offb, mskw, mskb, catw_h, catb);
  k_siren  <<<1, 256, 0, stream>>>(ls, knw1, knb1, knw2, knb2, knwo, knbo, kw, env);
  k_gemm_wmma<<<512, 256, 0, stream>>>(x_h,    inw_h,  inb,  xproj, NL_, C_, C_);
  k_dwln   <<<NL_, 256, 0, stream>>>(x, dww, dwb, lng, lnb, xdw_h);
  k_gemm_wmma<<<512, 256, 0, stream>>>(xdw_h,  catw_h, catb, rawom, NL_, C_, C_);
  k_softmax<<<NL_, 128, 0, stream>>>(rawom, env, offsb, attnb, accs);
  k_deform <<<NL_, 256, 0, stream>>>(xproj, offsb, attnb, kw, gout_h);
  k_gemm_wmma<<<512, 256, 0, stream>>>(gout_h, outw_h, outb, out,   NL_, C_, C_);
  k_semean <<<N_, 256, 0, stream>>>(out, colmean);
  k_semlp  <<<N_, 256, 0, stream>>>(colmean, sew1, seb1, sew2, seb2, sew);
  k_sescale<<<(NLC_+255)/256, 256, 0, stream>>>(out, sew);
  k_scalars<<<1, 32, 0, stream>>>(accs, out);
}
